// PatchCoreWRN50_23021024707268
// MI455X (gfx1250) — matrix operand order repaired and checked by reading, not compiled, not run on hardware
//
#include <hip/hip_runtime.h>
#include <stdint.h>

#define N_PATCH 1024
#define CH      3584
#define MROWS   131072
#define EPSN    1e-6f

typedef __bf16 v16bf __attribute__((ext_vector_type(16)));
typedef __bf16 v2bf  __attribute__((ext_vector_type(2)));
typedef float  v8f   __attribute__((ext_vector_type(8)));

union V16 {
    v16bf    v;
    uint32_t u[8];
    uint4    q[2];
};

// f32 pair -> packed bf16 (RNE). Prefer the hardware packed convert.
__device__ __forceinline__ uint32_t pk2(float lo, float hi) {
#if __has_builtin(__builtin_amdgcn_cvt_pk_bf16_f32)
    union { v2bf v; uint32_t u; } c;
    c.v = __builtin_amdgcn_cvt_pk_bf16_f32(lo, hi);
    return c.u;
#else
    union { v2bf v; uint32_t u; } c;
    c.v[0] = (__bf16)lo;
    c.v[1] = (__bf16)hi;
    return c.u;
#endif
}

// monotone order-preserving float -> uint encoding (for atomicMax on signed floats)
__device__ __forceinline__ uint32_t encOrd(float x) {
    uint32_t u = __float_as_uint(x);
    return (u & 0x80000000u) ? ~u : (u | 0x80000000u);
}
__device__ __forceinline__ float decOrd(uint32_t e) {
    return (e & 0x80000000u) ? __uint_as_float(e ^ 0x80000000u) : __uint_as_float(~e);
}

// ---------------------------------------------------------------- init
__global__ void pc_init_pmax(uint32_t* __restrict__ pmax) {
    int i = blockIdx.x * blockDim.x + threadIdx.x;
    if (i < N_PATCH) pmax[i] = 0u;   // below encOrd of any finite float
}

// ------------------------------------------------- patch-feat normalize -> bf16
// one 256-thread block per row; float2 loads; packed bf16 stores
__global__ void pc_norm_pf(const float* __restrict__ pf, uint32_t* __restrict__ pfn32) {
    __shared__ float red[8];
    const int row = blockIdx.x;
    const int tid = threadIdx.x;
    const float2* r2 = (const float2*)(pf + (size_t)row * CH);
    float s = 0.f;
#pragma unroll
    for (int i = 0; i < CH / 512; ++i) {           // 7 iters of float2
        float2 x = r2[tid + i * 256];
        s += x.x * x.x + x.y * x.y;
    }
    for (int m = 16; m >= 1; m >>= 1) s += __shfl_xor(s, m);
    if ((tid & 31) == 0) red[tid >> 5] = s;
    __syncthreads();
    if (tid < 8) {
        float t = red[tid];
        for (int m = 4; m >= 1; m >>= 1) t += __shfl_xor(t, m);
        if (tid == 0) red[0] = 1.f / (sqrtf(t) + EPSN);
    }
    __syncthreads();
    const float inv = red[0];
    uint32_t* o = pfn32 + (size_t)row * (CH / 2);
#pragma unroll
    for (int i = 0; i < CH / 512; ++i) {
        float2 x = r2[tid + i * 256];
        o[tid + i * 256] = pk2(x.x * inv, x.y * inv);
    }
}

// ------------------------------------------------- memory row inverse norms
__global__ void pc_norm_mem(const float* __restrict__ mem, float* __restrict__ invn) {
    __shared__ float red[8];
    const int row = blockIdx.x;
    const int tid = threadIdx.x;
    const float2* r2 = (const float2*)(mem + (size_t)row * CH);
    float s = 0.f;
#pragma unroll
    for (int i = 0; i < CH / 512; ++i) {
        float2 x = r2[tid + i * 256];
        s += x.x * x.x + x.y * x.y;
    }
    for (int m = 16; m >= 1; m >>= 1) s += __shfl_xor(s, m);
    if ((tid & 31) == 0) red[tid >> 5] = s;
    __syncthreads();
    if (tid < 8) {
        float t = red[tid];
        for (int m = 4; m >= 1; m >>= 1) t += __shfl_xor(t, m);
        if (tid == 0) invn[row] = 1.f / (sqrtf(t) + EPSN);
    }
}

// ------------------------------------------------- bf16 WMMA GEMM + row-max
// grid: (MROWS/64, N_PATCH/512), block: 256 (8 waves: 4 patch-groups x 2 mem-groups)
// wave tile: 128 patches x 32 mem rows = 8 x 2 accumulator tiles of 16x16.
__device__ __forceinline__ void loadcvtB(const float* __restrict__ row, V16& b) {
    const float4* p = (const float4*)row;
    float4 f0 = p[0], f1 = p[1], f2 = p[2], f3 = p[3];
    b.u[0] = pk2(f0.x, f0.y); b.u[1] = pk2(f0.z, f0.w);
    b.u[2] = pk2(f1.x, f1.y); b.u[3] = pk2(f1.z, f1.w);
    b.u[4] = pk2(f2.x, f2.y); b.u[5] = pk2(f2.z, f2.w);
    b.u[6] = pk2(f3.x, f3.y); b.u[7] = pk2(f3.z, f3.w);
}

__launch_bounds__(256)
__global__ void pc_gemm_max(const float* __restrict__ mem,
                            const uint16_t* __restrict__ pfn,
                            const float* __restrict__ invn,
                            uint32_t* __restrict__ pmax) {
    const int tid    = threadIdx.x;
    const int wave   = tid >> 5;
    const int lane   = tid & 31;
    const int hs     = lane >> 4;     // half-wave select
    const int ln     = lane & 15;
    const int waveP  = wave & 3;      // patch group 0..3
    const int waveM  = wave >> 2;     // mem group 0..1
    const int pBase  = blockIdx.y * 512 + waveP * 128;
    const int mBase  = blockIdx.x * 64  + waveM * 32;

    v8f acc[8][2];
    const v8f vzero = {0.f, 0.f, 0.f, 0.f, 0.f, 0.f, 0.f, 0.f};
#pragma unroll
    for (int s = 0; s < 8; ++s) { acc[s][0] = vzero; acc[s][1] = vzero; }

    // B source rows (f32, unnormalized): column n = ln maps to mem row mBase+g*16+ln
    const float* bRow0 = mem + (size_t)(mBase +      ln) * CH;
    const float* bRow1 = mem + (size_t)(mBase + 16 + ln) * CH;
    // A source row base (bf16 normalized): row = pBase + s*16 + ln
    const uint16_t* aBase = pfn + (size_t)(pBase + ln) * CH;

    for (int k0 = 0; k0 < CH; k0 += 32) {
        // B fragments: lane holds K = k0 + hs*16 + i (i = 0..15); hw cvt f32->bf16
        V16 b0, b1;
        loadcvtB(bRow0 + k0 + hs * 16, b0);
        loadcvtB(bRow1 + k0 + hs * 16, b1);
        // A fragments: lane holds K = k0 + hs*16 + i (i = 0..15)
        const uint16_t* aK = aBase + k0 + hs * 16;
#pragma unroll
        for (int s = 0; s < 8; ++s) {
            V16 a;
            const uint16_t* ar = aK + (size_t)s * 16 * CH;
            a.q[0] = *(const uint4*)(ar);
            a.q[1] = *(const uint4*)(ar + 8);
            acc[s][0] = __builtin_amdgcn_wmma_f32_16x16x32_bf16(
                false, a.v, false, b0.v, (short)0, acc[s][0], false, false);
            acc[s][1] = __builtin_amdgcn_wmma_f32_16x16x32_bf16(
                false, a.v, false, b1.v, (short)0, acc[s][1], false, false);
        }
    }

    // fold memory-row normalization into the max; reduce N (16 lanes) by butterfly
    const float inv0 = invn[mBase +      ln];
    const float inv1 = invn[mBase + 16 + ln];
#pragma unroll
    for (int s = 0; s < 8; ++s) {
#pragma unroll
        for (int v = 0; v < 8; ++v) {
            float x = fmaxf(acc[s][0][v] * inv0, acc[s][1][v] * inv1);
            x = fmaxf(x, __shfl_xor(x, 1));
            x = fmaxf(x, __shfl_xor(x, 2));
            x = fmaxf(x, __shfl_xor(x, 4));
            x = fmaxf(x, __shfl_xor(x, 8));
            // C layout: VGPR v, lanes 0-15 -> M=v, lanes 16-31 -> M=v+8
            if (ln == v)
                atomicMax(&pmax[pBase + s * 16 + v + 8 * hs], encOrd(x));
        }
    }
}

// ------------------------------------------------- scores + top-k mean
__global__ void pc_finalize(const uint32_t* __restrict__ pmax,
                            const int* __restrict__ topk,
                            float* __restrict__ out) {
    __shared__ float sc[N_PATCH];
    __shared__ float rv[N_PATCH];
    __shared__ int   ri[N_PATCH];
    __shared__ float accum;
    const int tid = threadIdx.x;
    float score = 1.0f - decOrd(pmax[tid]);
    out[tid] = score;
    sc[tid]  = score;
    if (tid == 0) accum = 0.f;
    __syncthreads();
    int k = *topk;
    if (k > N_PATCH) k = N_PATCH;
    if (k < 1) k = 1;
    for (int t = 0; t < k; ++t) {
        rv[tid] = sc[tid]; ri[tid] = tid;
        __syncthreads();
        for (int off = 512; off > 0; off >>= 1) {
            if (tid < off) {
                if (rv[tid + off] > rv[tid]) { rv[tid] = rv[tid + off]; ri[tid] = ri[tid + off]; }
            }
            __syncthreads();
        }
        if (tid == 0) { accum += rv[0]; sc[ri[0]] = -INFINITY; }
        __syncthreads();
    }
    if (tid == 0) out[N_PATCH] = accum / (float)k;
}

// ---------------------------------------------------------------- launch
extern "C" void kernel_launch(void* const* d_in, const int* in_sizes, int n_in,
                              void* d_out, int out_size, void* d_ws, size_t ws_size,
                              hipStream_t stream) {
    (void)in_sizes; (void)n_in; (void)out_size; (void)ws_size;
    const float* pf   = (const float*)d_in[0];
    const float* mem  = (const float*)d_in[1];
    const int*   topk = (const int*)d_in[2];
    float*       out  = (float*)d_out;

    uint8_t*  ws    = (uint8_t*)d_ws;
    uint32_t* pmax  = (uint32_t*)ws;                                 //   4 KB
    float*    invn  = (float*)(ws + 4096);                           // 512 KB
    uint32_t* pfn32 = (uint32_t*)(ws + 4096 + (size_t)MROWS * 4);    //   7 MB
    uint16_t* pfn   = (uint16_t*)pfn32;

    pc_init_pmax<<<4, 256, 0, stream>>>(pmax);
    pc_norm_pf  <<<N_PATCH, 256, 0, stream>>>(pf, pfn32);
    pc_norm_mem <<<MROWS, 256, 0, stream>>>(mem, invn);
    dim3 grid(MROWS / 64, N_PATCH / 512);
    pc_gemm_max <<<grid, 256, 0, stream>>>(mem, pfn, invn, pmax);
    pc_finalize <<<1, 1024, 0, stream>>>(pmax, topk, out);
}
